// UFCLayer_88270167867864
// MI455X (gfx1250) — compile-verified, matrix operand order checked by reading, not run on hardware
//
#include <hip/hip_runtime.h>
#include <hip/hip_bf16.h>
#include <math.h>

// ---------------------------------------------------------------------------
// MI455X (gfx1250) implementation of UFCLayer forward.
// All matrix contractions (linears, attention, correlation Grams, 4D convs as
// implicit GEMM) run on v_wmma_f32_16x16x32_f16 (fp16 in, fp32 accum).
// GEMM operand tiles are staged into LDS via the Tensor Data Mover
// (tensor_load_to_lds + s_wait_tensorcnt, double-buffered) when available.
// Param pointer order follows JAX pytree flattening (dicts sorted by key).
// ---------------------------------------------------------------------------

typedef _Float16 half_t;
typedef __attribute__((ext_vector_type(16))) _Float16 v16h;
typedef __attribute__((ext_vector_type(8)))  _Float16 v8h;
typedef __attribute__((ext_vector_type(8)))  float    v8f;
typedef __attribute__((ext_vector_type(4)))  unsigned int v4u;
typedef __attribute__((ext_vector_type(4)))  int v4i;
typedef __attribute__((ext_vector_type(8)))  int v8i;

#if __has_builtin(__builtin_amdgcn_tensor_load_to_lds) && \
    __has_builtin(__builtin_amdgcn_s_wait_tensorcnt)
#define HAVE_TDM 1
#else
#define HAVE_TDM 0
#endif

#define NB 16
#define NHD 4
#define CORR_N ((size_t)NB * NHD * 256 * 256)   // 4194304
#define FEAT_N ((size_t)NB * 256 * 256)         // 1048576

union FragU { v16h v; v8h h[2]; half_t e[16]; };

// Load a 16x32 fp16 fragment from a row-major [rows x ld] matrix in global.
// ISA layout: lanes 0-15 -> row, kb=0; lanes 16-31 -> row, kb=8.
// halves 0-7 = K kb..kb+7 ; halves 8-15 = K kb+16..kb+23 (two 16B loads).
__device__ __forceinline__ v16h load_frag(const half_t* base, int row, int ld,
                                          int k0, int lane) {
  int kb = k0 + ((lane & 16) ? 8 : 0);
  const half_t* p = base + (size_t)row * ld + kb;
  FragU f;
  f.h[0] = *(const v8h*)(p);
  f.h[1] = *(const v8h*)(p + 16);
  return f.v;
}

// Same fragment load but from an LDS-resident [rows x 32] tile.
__device__ __forceinline__ v16h lds_frag(const half_t* ls, int row, int lane) {
  int kb = (lane & 16) ? 8 : 0;
  const half_t* p = ls + row * 32 + kb;
  FragU f;
  f.h[0] = *(const v8h*)(p);
  f.h[1] = *(const v8h*)(p + 16);
  return f.v;
}

// Issue one TDM 2D tile load: rows x 32 fp16 elements, row stride ld
// (elements), from global `g` into LDS byte offset lds_off.
// D# per CDNA5 ISA ch.8: group0 = {count=1, lds_addr, global_addr, type=2};
// group1 = {data_size=2B, tensor_dim0=32, tensor_dim1=rows, tile_dim0=32,
//           tile_dim1=rows, dim0_stride=ld}.
__device__ __forceinline__ void tdm_load_2d(const half_t* g, unsigned lds_off,
                                            int rows, int ld) {
#if HAVE_TDM
  unsigned long long ga = (unsigned long long)(size_t)g;
  v4u g0 = {1u, lds_off, (unsigned)ga, (unsigned)(ga >> 32) | (2u << 30)};
  v8i g1 = {(int)0x00010000,        // data_size=1 (2 bytes), no flags, no mask
            (int)(32u << 16),        // tensor_dim0[15:0] = 32
            (int)((unsigned)rows << 16), // tensor_dim1[15:0] = rows
            (int)(32u << 16),        // tile_dim0 = 32
            rows,                    // tile_dim1 = rows
            ld,                      // tensor_dim0_stride (elements)
            0, 0};
  v4i z4 = {0, 0, 0, 0};
#if __clang_major__ >= 23
  v8i z8 = {0, 0, 0, 0, 0, 0, 0, 0};
  __builtin_amdgcn_tensor_load_to_lds(g0, g1, z4, z4, z8, 0);
#else
  __builtin_amdgcn_tensor_load_to_lds(g0, g1, z4, z4, 0);
#endif
#else
  (void)g; (void)lds_off; (void)rows; (void)ld;
#endif
}

// ------------------------------- GEMM --------------------------------------
// D[M,N] (f32 and/or f16) = A[M,K](f16,rowmajor) * W[N,K](f16,rowmajor)^T
//                           + bias[n] + residual[m,n]
// One wave computes a 32x64 tile (2x4 WMMA tiles -> 0.75 frag loads / wmma).
// batched: z -> (b = z / H, h = z % H); per-operand (b,h) strides.
// Requires M%32==0, N%64==0, K%32==0 (all call sites satisfy this).
struct GemmP {
  const half_t* A; long sAb, sAh; int lda;
  const half_t* W; long sWb, sWh; int ldw;
  const float* bias;
  const float* R; long sRb, sRh; int ldr;
  float* D; long sDb, sDh; int ldd;
  half_t* Dh; long sDhb, sDhh; int lddh;
  int K, H;
};

__global__ __launch_bounds__(32) void k_wmma_gemm(GemmP g) {
  const int lane = threadIdx.x;
  const int m0 = blockIdx.x * 32, n0 = blockIdx.y * 64;
  const int z = blockIdx.z, b = z / g.H, h = z % g.H;
  const half_t* A = g.A + (long)b * g.sAb + (long)h * g.sAh;
  const half_t* W = g.W + (long)b * g.sWb + (long)h * g.sWh;
  v8f acc[2][4] = {};

#if HAVE_TDM
  // Double-buffered TDM pipeline: LDS tiles A:32x32, W:64x32 (fp16).
  __shared__ __align__(16) half_t lsA[2][32 * 32];
  __shared__ __align__(16) half_t lsW[2][64 * 32];
  const int nchunks = g.K / 32;
  const half_t* Abase = A + (size_t)m0 * g.lda;
  const half_t* Wbase = W + (size_t)n0 * g.ldw;
  tdm_load_2d(Abase, (unsigned)(size_t)&lsA[0][0], 32, g.lda);
  tdm_load_2d(Wbase, (unsigned)(size_t)&lsW[0][0], 64, g.ldw);
  for (int c2 = 0; c2 < nchunks; ++c2) {
    const int buf = c2 & 1;
    if (c2 + 1 < nchunks) {
      const int k0n = (c2 + 1) * 32;
      tdm_load_2d(Abase + k0n, (unsigned)(size_t)&lsA[buf ^ 1][0], 32, g.lda);
      tdm_load_2d(Wbase + k0n, (unsigned)(size_t)&lsW[buf ^ 1][0], 64, g.ldw);
      __builtin_amdgcn_s_wait_tensorcnt((short)2);   // chunk c2 has landed
    } else {
      __builtin_amdgcn_s_wait_tensorcnt((short)0);
    }
    v16h a[2], w[4];
#pragma unroll
    for (int i = 0; i < 2; ++i)
      a[i] = lds_frag(&lsA[buf][0], i * 16 + (lane & 15), lane);
#pragma unroll
    for (int j = 0; j < 4; ++j)
      w[j] = lds_frag(&lsW[buf][0], j * 16 + (lane & 15), lane);
#pragma unroll
    for (int i = 0; i < 2; ++i)
#pragma unroll
      for (int j = 0; j < 4; ++j)
        acc[i][j] = __builtin_amdgcn_wmma_f32_16x16x32_f16(
            false, a[i], false, w[j], (short)0, acc[i][j], false, false);
  }
#else
  for (int k0 = 0; k0 < g.K; k0 += 32) {
    __builtin_prefetch(A + (size_t)(m0 + (lane & 15)) * g.lda + k0 + 64, 0, 1);
    v16h a[2], w[4];
#pragma unroll
    for (int i = 0; i < 2; ++i)
      a[i] = load_frag(A, m0 + i * 16 + (lane & 15), g.lda, k0, lane);
#pragma unroll
    for (int j = 0; j < 4; ++j)
      w[j] = load_frag(W, n0 + j * 16 + (lane & 15), g.ldw, k0, lane);
#pragma unroll
    for (int i = 0; i < 2; ++i)
#pragma unroll
      for (int j = 0; j < 4; ++j)
        acc[i][j] = __builtin_amdgcn_wmma_f32_16x16x32_f16(
            false, a[i], false, w[j], (short)0, acc[i][j], false, false);
  }
#endif

  const int nl = lane & 15;
  const int ml = (lane & 16) ? 8 : 0;
  const float* Rb = g.R ? (g.R + (long)b * g.sRb + (long)h * g.sRh) : nullptr;
  float* Db = g.D ? (g.D + (long)b * g.sDb + (long)h * g.sDh) : nullptr;
  half_t* Dhb = g.Dh ? (g.Dh + (long)b * g.sDhb + (long)h * g.sDhh) : nullptr;
#pragma unroll
  for (int j = 0; j < 4; ++j) {
    const int nn = n0 + j * 16 + nl;
    const float bias = g.bias ? g.bias[nn] : 0.0f;
#pragma unroll
    for (int i = 0; i < 2; ++i) {
#pragma unroll
      for (int r = 0; r < 8; ++r) {
        const int mm = m0 + i * 16 + ml + r;
        float v = acc[i][j][r] + bias;
        if (Rb) v += Rb[(size_t)mm * g.ldr + nn];
        if (Db) Db[(size_t)mm * g.ldd + nn] = v;
        if (Dhb) Dhb[(size_t)mm * g.lddh + nn] = (half_t)v;
      }
    }
  }
}

// ------------------------------ conv4d -------------------------------------
// Implicit GEMM: D[co, 16 tx-positions] accumulated over K = Ci*81 (zero-pad
// to Kpad mult-of-32; Co zero-padded to 16 in prepped weights).
// One wave per tile = fixed (b, sy, sx, ty), tx = lane&15.
__global__ __launch_bounds__(32) void k_conv4d(const half_t* __restrict__ X,
                                               const half_t* __restrict__ Wh,
                                               const float* __restrict__ bias,
                                               float* __restrict__ Y,
                                               int Ci, int Co, int Kpad) {
  const int lane = threadIdx.x;
  const int tid = blockIdx.x;           // NB*16*16*16
  const int b = tid >> 12;
  const int sy = (tid >> 8) & 15;
  const int sx = (tid >> 4) & 15;
  const int ty = tid & 15;
  const int tx = lane & 15;
  const int kreal = Ci * 81;
  const int kb_lane = (lane & 16) ? 8 : 0;
  v8f acc = {};
  for (int k0 = 0; k0 < Kpad; k0 += 32) {
    v16h a = load_frag(Wh, lane & 15, Kpad, k0, lane);
    FragU bf;
#pragma unroll
    for (int j = 0; j < 16; ++j) {
      int kj = k0 + kb_lane + ((j < 8) ? j : (j + 8));
      half_t v = (half_t)0.0f;
      if (kj < kreal) {
        int ci = kj / 81;
        int r = kj - ci * 81;
        int dy = r / 27;  int r2 = r - dy * 27;
        int dx = r2 / 9;  int r3 = r2 - dx * 9;
        int ey = r3 / 3;  int ex = r3 - ey * 3;
        int syi = sy + dy - 1, sxi = sx + dx - 1;
        int tyi = ty + ey - 1, txi = tx + ex - 1;
        if (((unsigned)syi < 16u) & ((unsigned)sxi < 16u) &
            ((unsigned)tyi < 16u) & ((unsigned)txi < 16u)) {
          v = X[(((size_t)(b * Ci + ci)) << 16) + (syi << 12) + (sxi << 8) +
                (tyi << 4) + txi];
        }
      }
      bf.e[j] = v;
    }
    acc = __builtin_amdgcn_wmma_f32_16x16x32_f16(false, a, false, bf.v,
                                                 (short)0, acc, false, false);
  }
  const int cob = (lane & 16) ? 8 : 0;
#pragma unroll
  for (int r = 0; r < 8; ++r) {
    int co = cob + r;
    if (co < Co)
      Y[(((size_t)(b * Co + co)) << 16) + (sy << 12) + (sx << 8) + (ty << 4) + tx] =
          acc[r] + bias[co];
  }
}

// --------------------------- elementwise / reductions ----------------------

__global__ void k_cvt(const float* __restrict__ s, half_t* __restrict__ d, size_t n) {
  size_t i = (size_t)blockIdx.x * blockDim.x + threadIdx.x;
  if (i < n) d[i] = (half_t)s[i];
}

__global__ void k_zero(float* p, int n) {
  int i = blockIdx.x * blockDim.x + threadIdx.x;
  if (i < n) p[i] = 0.0f;
}

__global__ void k_prep_convw(const float* __restrict__ w, half_t* __restrict__ wh,
                             int Ci, int Co, int Kpad) {
  int idx = blockIdx.x * blockDim.x + threadIdx.x;
  if (idx >= 16 * Kpad) return;
  int k = idx % Kpad, co = idx / Kpad;
  int kreal = Ci * 81;
  wh[idx] = (co < Co && k < kreal) ? (half_t)w[(size_t)co * kreal + k] : (half_t)0.0f;
}

__global__ __launch_bounds__(256) void k_layernorm(const float* __restrict__ x,
                                                   const float* __restrict__ gw,
                                                   const float* __restrict__ gb,
                                                   half_t* __restrict__ yh) {
  __shared__ float red[256];
  size_t row = blockIdx.x; int t = threadIdx.x;
  float v = x[row * 256 + t];
  red[t] = v; __syncthreads();
  for (int s2 = 128; s2 > 0; s2 >>= 1) { if (t < s2) red[t] += red[t + s2]; __syncthreads(); }
  float m = red[0] * (1.0f / 256.0f); __syncthreads();
  float d = v - m;
  red[t] = d * d; __syncthreads();
  for (int s2 = 128; s2 > 0; s2 >>= 1) { if (t < s2) red[t] += red[t + s2]; __syncthreads(); }
  float rs = rsqrtf(red[0] * (1.0f / 256.0f) + 1e-5f);
  yh[row * 256 + t] = (half_t)(d * rs * gw[t] + gb[t]);
}

__global__ void k_build_cf(const float* __restrict__ corr,
                           const half_t* __restrict__ featn,
                           half_t* __restrict__ cf) {
  size_t idx = (size_t)blockIdx.x * blockDim.x + threadIdx.x;
  if (idx >= (size_t)NB * 256 * 1280) return;
  int j = (int)(idx % 1280);
  size_t bs = idx / 1280;
  int s = (int)(bs & 255); int b = (int)(bs >> 8);
  if (j < 1024) {
    int h = j >> 8, t = j & 255;
    cf[idx] = (half_t)corr[(((size_t)(b * 4 + h)) << 16) + (s << 8) + t];
  } else {
    cf[idx] = featn[(((size_t)b << 8) + s) * 256 + (j - 1024)];
  }
}

__device__ __forceinline__ float elu1(float x) { return x > 0.f ? x + 1.f : expf(x); }

__global__ void k_q_post(const float* __restrict__ q, const float* __restrict__ pos,
                         half_t* __restrict__ Qh) {
  size_t idx = (size_t)blockIdx.x * blockDim.x + threadIdx.x;
  if (idx >= FEAT_N) return;
  int cch = (int)(idx & 255); size_t bl = idx >> 8;
  int l = (int)(bl & 255); int b = (int)(bl >> 8);
  int h = cch >> 6, d = cch & 63;
  float v = elu1(q[idx] + pos[l * 64 + d]);
  Qh[((((size_t)b * 4 + h) * 256 + l) << 6) + d] = (half_t)v;
}

__global__ void k_k_post(const float* __restrict__ k, const float* __restrict__ pos,
                         half_t* __restrict__ Kt) {
  size_t idx = (size_t)blockIdx.x * blockDim.x + threadIdx.x;
  if (idx >= FEAT_N) return;
  int cch = (int)(idx & 255); size_t bl = idx >> 8;
  int l = (int)(bl & 255); int b = (int)(bl >> 8);
  int h = cch >> 6, d = cch & 63;
  float v = elu1(k[idx] + pos[l * 64 + d]);
  Kt[(((size_t)b * 4 + h) * 64 + d) * 256 + l] = (half_t)v;
}

__global__ void k_rowsum(const half_t* __restrict__ Kt, float* __restrict__ Ksum) {
  int idx = blockIdx.x * blockDim.x + threadIdx.x;
  if (idx >= NB * NHD * 64) return;
  const half_t* p = Kt + (size_t)idx * 256;
  float s = 0.f;
  for (int i = 0; i < 256; ++i) s += (float)p[i];
  Ksum[idx] = s;
}

__global__ void k_z(const half_t* __restrict__ Qh, const float* __restrict__ Ksum,
                    float* __restrict__ Z) {
  int idx = blockIdx.x * blockDim.x + threadIdx.x;
  if (idx >= NB * NHD * 256) return;
  const half_t* q = Qh + ((size_t)idx << 6);
  const float* ks = Ksum + (size_t)(idx >> 8) * 64;
  float s = 0.f;
  for (int d = 0; d < 64; ++d) s += (float)q[d] * ks[d];
  Z[idx] = 1.0f / (s + 1e-6f);
}

__global__ void k_headT(const float* __restrict__ v, half_t* __restrict__ vt) {
  size_t idx = (size_t)blockIdx.x * blockDim.x + threadIdx.x;
  if (idx >= FEAT_N) return;
  int t = (int)(idx & 255); int cc = (int)((idx >> 8) & 63);
  int h = (int)((idx >> 14) & 3); int b = (int)(idx >> 16);
  vt[idx] = (half_t)v[((((size_t)b << 8) + t) << 8) + h * 64 + cc];
}

__global__ void k_apply_feat(const float* __restrict__ fr, const float* __restrict__ mp,
                             const float* __restrict__ Z, float* __restrict__ out) {
  size_t idx = (size_t)blockIdx.x * blockDim.x + threadIdx.x;
  if (idx >= FEAT_N) return;
  int cch = (int)(idx & 255); size_t bl = idx >> 8;
  int l = (int)(bl & 255); int b = (int)(bl >> 8);
  int h = cch >> 6, d = cch & 63;
  size_t bhl = ((size_t)b * 4 + h) * 256 + l;
  out[idx] = fr[idx] + mp[(bhl << 6) + d] * Z[bhl];
}

__global__ void k_apply_corr(const float* __restrict__ cin, const float* __restrict__ mp,
                             const float* __restrict__ Z, float* __restrict__ out,
                             half_t* __restrict__ outh) {
  size_t idx = (size_t)blockIdx.x * blockDim.x + threadIdx.x;
  if (idx >= CORR_N) return;
  float v = cin[idx] + mp[idx] * Z[idx >> 8];
  out[idx] = v;
  if (outh) outh[idx] = (half_t)v;
}

__global__ void k_transpose_st(const float* __restrict__ src, float* __restrict__ df,
                               half_t* __restrict__ dh) {
  size_t idx = (size_t)blockIdx.x * blockDim.x + threadIdx.x;
  if (idx >= CORR_N) return;
  int s = (int)(idx & 255); int t = (int)((idx >> 8) & 255);
  size_t z = idx >> 16;
  float v = src[(z << 16) + ((size_t)s << 8) + t];
  if (df) df[idx] = v;
  if (dh) dh[idx] = (half_t)v;
}

__global__ void k_add(const float* __restrict__ a, const float* __restrict__ b,
                      float* __restrict__ d, half_t* __restrict__ dh, size_t n) {
  size_t i = (size_t)blockIdx.x * blockDim.x + threadIdx.x;
  if (i >= n) return;
  float v = a[i] + b[i];
  d[i] = v;
  if (dh) dh[i] = (half_t)v;
}

__global__ void k_add_transposed(const float* __restrict__ a, const float* __restrict__ bT,
                                 float* __restrict__ d) {
  size_t idx = (size_t)blockIdx.x * blockDim.x + threadIdx.x;
  if (idx >= CORR_N) return;
  int t = (int)(idx & 255); int s = (int)((idx >> 8) & 255);
  size_t z = idx >> 16;
  d[idx] = a[idx] + bT[(z << 16) + ((size_t)t << 8) + s];
}

__global__ void k_dwconv_gelu(const float* __restrict__ hid, const float* __restrict__ dww,
                              const float* __restrict__ dwb, half_t* __restrict__ out,
                              int C) {
  size_t idx = (size_t)blockIdx.x * blockDim.x + threadIdx.x;
  if (idx >= (size_t)NB * 256 * C) return;
  int cc = (int)(idx % C);
  size_t bl = idx / C;
  int l = (int)(bl & 255); int b = (int)(bl >> 8);
  int yy = l >> 4, xx = l & 15;
  float s = dwb[cc];
#pragma unroll
  for (int dy = 0; dy < 3; ++dy)
#pragma unroll
    for (int dx = 0; dx < 3; ++dx) {
      int yi = yy + dy - 1, xi = xx + dx - 1;
      if (((unsigned)yi < 16u) & ((unsigned)xi < 16u))
        s += dww[cc * 9 + dy * 3 + dx] *
             hid[(((size_t)b << 8) + (yi * 16 + xi)) * C + cc];
    }
  out[idx] = (half_t)(0.5f * s * (1.0f + erff(s * 0.70710678118f)));
}

__global__ __launch_bounds__(256) void k_gn_reduce(const float* __restrict__ x,
                                                   float* __restrict__ acc, int perB) {
  __shared__ float s1[256], s2[256];
  size_t idx = (size_t)blockIdx.x * 256 + threadIdx.x;
  float v = x[idx];
  s1[threadIdx.x] = v; s2[threadIdx.x] = v * v; __syncthreads();
  for (int st = 128; st > 0; st >>= 1) {
    if (threadIdx.x < st) { s1[threadIdx.x] += s1[threadIdx.x + st];
                            s2[threadIdx.x] += s2[threadIdx.x + st]; }
    __syncthreads();
  }
  if (threadIdx.x == 0) {
    int b = (int)(idx / (size_t)perB);
    atomicAdd(&acc[b * 2], s1[0]);
    atomicAdd(&acc[b * 2 + 1], s2[0]);
  }
}

__global__ void k_gn_relu(const float* __restrict__ x, const float* __restrict__ acc,
                          const float* __restrict__ gg, const float* __restrict__ gb,
                          float* __restrict__ yf, half_t* __restrict__ yh, int Co) {
  size_t perB = (size_t)Co << 16;
  size_t idx = (size_t)blockIdx.x * blockDim.x + threadIdx.x;
  if (idx >= (size_t)NB * perB) return;
  int b = (int)(idx / perB);
  int c = (int)((idx >> 16) % Co);
  float cnt = (float)perB;
  float m = acc[b * 2] / cnt;
  float var = acc[b * 2 + 1] / cnt - m * m;
  float v = (x[idx] - m) * rsqrtf(var + 1e-5f) * gg[c] + gb[c];
  v = v > 0.f ? v : 0.f;
  if (yf) yf[idx] = v;
  if (yh) yh[idx] = (half_t)v;
}

__global__ __launch_bounds__(256) void k_softmax_row(const float* __restrict__ x,
                                                     half_t* __restrict__ p) {
  __shared__ float red[256];
  size_t row = blockIdx.x; int t = threadIdx.x;
  float v = x[(row << 8) + t];
  red[t] = v; __syncthreads();
  for (int st = 128; st > 0; st >>= 1) { if (t < st) red[t] = fmaxf(red[t], red[t + st]); __syncthreads(); }
  float mx = red[0]; __syncthreads();
  float e = expf(v - mx);
  red[t] = e; __syncthreads();
  for (int st = 128; st > 0; st >>= 1) { if (t < st) red[t] += red[t + st]; __syncthreads(); }
  p[(row << 8) + t] = (half_t)(e / red[0]);
}

__global__ __launch_bounds__(256) void k_softmax_col(const float* __restrict__ x,
                                                     half_t* __restrict__ p) {
  __shared__ float red[256];
  size_t zt = blockIdx.x; int s = threadIdx.x;
  size_t z = zt >> 8; int t = (int)(zt & 255);
  float v = x[(z << 16) + ((size_t)s << 8) + t];
  red[s] = v; __syncthreads();
  for (int st = 128; st > 0; st >>= 1) { if (s < st) red[s] = fmaxf(red[s], red[s + st]); __syncthreads(); }
  float mx = red[0]; __syncthreads();
  float e = expf(v - mx);
  red[s] = e; __syncthreads();
  for (int st = 128; st > 0; st >>= 1) { if (s < st) red[s] += red[s + st]; __syncthreads(); }
  p[(z << 16) + ((size_t)t << 8) + s] = (half_t)(e / red[0]);
}

__global__ __launch_bounds__(256) void k_colnorm(const float* __restrict__ x,
                                                 half_t* __restrict__ nh) {
  __shared__ float red[256];
  size_t row = blockIdx.x; int c = threadIdx.x;
  float v = x[(row << 8) + c];
  red[c] = v * v; __syncthreads();
  for (int st = 128; st > 0; st >>= 1) { if (c < st) red[c] += red[c + st]; __syncthreads(); }
  float inv = 1.0f / (sqrtf(red[0]) + 1e-5f);
  nh[(row << 8) + c] = (half_t)(v * inv);
}

// ------------------------------ host side ----------------------------------

static inline dim3 blocks_for(size_t n, int bs) {
  return dim3((unsigned int)((n + (size_t)bs - 1) / (size_t)bs));
}

struct ConvLayer { const half_t* wh; const float* bias; const float* gn_g;
                   const float* gn_b; int Ci, Co, Kpad; };

struct Ctx {
  hipStream_t st;
  const float* in[68];
  half_t *wq, *wk, *wv, *wvx, *l1w, *l2w, *xl1w, *xl2w;
  ConvLayer Lvpc, Lmc0, Lmc1, Lmr0, Lmr1, Lm2a, Lm2b, Lf1, Lf2;
  half_t *lnA, *cf, *Qh, *Kt, *Vft, *Vct, *KVtf, *KVtc, *corrMid_h, *hidact, *hid16h;
  half_t *snorm, *tnorm, *tok_h, *Prow, *Pcol, *vSt, *vTt, *corrR2h, *corrR4h;
  float *qraw, *kraw, *vraw, *Ksum, *Zb, *msgf, *msgc, *featMid, *corrMid;
  float *hid, *t4a, *t4b, *hid16, *acc, *tok, *vS_f, *vT_f;
};

static GemmP gp(const half_t* A, long sAb, long sAh, int lda,
                const half_t* W, long sWb, long sWh, int ldw,
                const float* bias,
                const float* R, long sRb, long sRh, int ldr,
                float* D, long sDb, long sDh, int ldd,
                half_t* Dh, long sDhb, long sDhh, int lddh,
                int K, int H) {
  GemmP p{A, sAb, sAh, lda, W, sWb, sWh, ldw, bias, R, sRb, sRh, ldr,
          D, sDb, sDh, ldd, Dh, sDhb, sDhh, lddh, K, H};
  return p;
}

static void launch_gemm(hipStream_t st, const GemmP& p, int M, int N, int batch) {
  k_wmma_gemm<<<dim3(M / 32, N / 64, batch), dim3(32), 0, st>>>(p);
}

static void conv_gn_relu(Ctx& c, const ConvLayer& Lr, const half_t* Xh,
                         float* raw, float* Yf, half_t* Yh) {
  k_conv4d<<<dim3(NB * 16 * 16 * 16), dim3(32), 0, c.st>>>(Xh, Lr.wh, Lr.bias, raw,
                                                           Lr.Ci, Lr.Co, Lr.Kpad);
  size_t n = (size_t)NB * Lr.Co * 65536;
  k_zero<<<dim3(1), dim3(64), 0, c.st>>>(c.acc, 64);
  k_gn_reduce<<<blocks_for(n, 256), dim3(256), 0, c.st>>>(raw, c.acc, Lr.Co << 16);
  k_gn_relu<<<blocks_for(n, 256), dim3(256), 0, c.st>>>(raw, c.acc, Lr.gn_g, Lr.gn_b,
                                                        Yf, Yh, Lr.Co);
}

static void run_mlp(Ctx& c, const float* ln_b, const float* ln_g,
                    const half_t* w1, const float* b1,
                    const float* dww, const float* dwb,
                    const half_t* w2, const float* b2,
                    const float* xin, float* xout) {
  k_layernorm<<<dim3(NB * 256), dim3(256), 0, c.st>>>(xin, ln_g, ln_b, c.lnA);
  launch_gemm(c.st, gp(c.lnA, 0, 0, 256, w1, 0, 0, 256, b1,
                       nullptr, 0, 0, 0, c.hid, 0, 0, 1024,
                       nullptr, 0, 0, 0, 256, 1), 4096, 1024, 1);
  k_dwconv_gelu<<<blocks_for((size_t)NB * 256 * 1024, 256), dim3(256), 0, c.st>>>(
      c.hid, dww, dwb, c.hidact, 1024);
  launch_gemm(c.st, gp(c.hidact, 0, 0, 1024, w2, 0, 0, 1024, b2,
                       xin, 0, 0, 256, xout, 0, 0, 256,
                       nullptr, 0, 0, 0, 1024, 1), 4096, 256, 1);
}

// forward_attention branch (corr in branch-domain [B,4,s,t])
static void run_fa(Ctx& c, const float* corrF, const half_t* corrH,
                   const float* featF, float* corrOut, float* featOut) {
  hipStream_t st = c.st;
  // norm1 + cf = [corr_cat | featn]
  k_layernorm<<<dim3(NB * 256), dim3(256), 0, st>>>(featF, c.in[48], c.in[47], c.lnA);
  k_build_cf<<<blocks_for((size_t)NB * 256 * 1280, 256), dim3(256), 0, st>>>(
      corrF, c.lnA, c.cf);
  // q,k projections (K=1280)
  launch_gemm(st, gp(c.cf, 0, 0, 1280, c.wq, 0, 0, 1280, c.in[56],
                     nullptr, 0, 0, 0, c.qraw, 0, 0, 256,
                     nullptr, 0, 0, 0, 1280, 1), 4096, 256, 1);
  launch_gemm(st, gp(c.cf, 0, 0, 1280, c.wk, 0, 0, 1280, c.in[9],
                     nullptr, 0, 0, 0, c.kraw, 0, 0, 256,
                     nullptr, 0, 0, 0, 1280, 1), 4096, 256, 1);
  k_q_post<<<blocks_for(FEAT_N, 256), dim3(256), 0, st>>>(c.qraw, c.in[55], c.Qh);
  k_k_post<<<blocks_for(FEAT_N, 256), dim3(256), 0, st>>>(c.kraw, c.in[55], c.Kt);
  k_rowsum<<<blocks_for(4096, 256), dim3(256), 0, st>>>(c.Kt, c.Ksum);
  k_z<<<blocks_for(16384, 256), dim3(256), 0, st>>>(c.Qh, c.Ksum, c.Zb);
  // v_feat
  launch_gemm(st, gp(c.lnA, 0, 0, 256, c.wv, 0, 0, 256, c.in[60],
                     nullptr, 0, 0, 0, c.vraw, 0, 0, 256,
                     nullptr, 0, 0, 0, 256, 1), 4096, 256, 1);
  k_headT<<<blocks_for(FEAT_N, 256), dim3(256), 0, st>>>(c.vraw, c.Vft);
  // v_corr = relu(GN(conv4d_4to4(corr)))
  conv_gn_relu(c, c.Lvpc, corrH, c.t4a, c.t4b, nullptr);
  k_transpose_st<<<blocks_for(CORR_N, 256), dim3(256), 0, st>>>(c.t4b, nullptr, c.Vct);
  // KV^T (note 1/S and *S of the reference cancel exactly)
  launch_gemm(st, gp(c.Vft, 16384, 0, 256, c.Kt, 16384, 0, 256, nullptr,
                     nullptr, 0, 0, 0, nullptr, 0, 0, 0,
                     c.KVtf, 4096, 0, 64, 256, 1), 64, 64, 64);
  launch_gemm(st, gp(c.Vct, 65536, 0, 256, c.Kt, 16384, 0, 256, nullptr,
                     nullptr, 0, 0, 0, nullptr, 0, 0, 0,
                     c.KVtc, 16384, 0, 64, 256, 1), 256, 64, 64);
  // msg = Q * KV, then *Z with residual
  launch_gemm(st, gp(c.Qh, 16384, 0, 64, c.KVtf, 4096, 0, 64, nullptr,
                     nullptr, 0, 0, 0, c.msgf, 16384, 0, 64,
                     nullptr, 0, 0, 0, 64, 1), 256, 64, 64);
  launch_gemm(st, gp(c.Qh, 16384, 0, 64, c.KVtc, 16384, 0, 64, nullptr,
                     nullptr, 0, 0, 0, c.msgc, 65536, 0, 256,
                     nullptr, 0, 0, 0, 64, 1), 256, 256, 64);
  k_apply_feat<<<blocks_for(FEAT_N, 256), dim3(256), 0, st>>>(featF, c.msgf, c.Zb,
                                                              c.featMid);
  k_apply_corr<<<blocks_for(CORR_N, 256), dim3(256), 0, st>>>(corrF, c.msgc, c.Zb,
                                                              c.corrMid, c.corrMid_h);
  // feat MLP (norm2 + mlp)
  run_mlp(c, c.in[49], c.in[50], c.l1w, c.in[13], c.in[12], c.in[11],
          c.l2w, c.in[15], c.featMid, featOut);
  // corr encoder (mlp_corr: 4->16->4) with residual
  conv_gn_relu(c, c.Lmc0, c.corrMid_h, c.hid16, nullptr, c.hid16h);
  conv_gn_relu(c, c.Lmc1, c.hid16h, c.t4a, c.t4b, nullptr);
  k_add<<<blocks_for(CORR_N, 256), dim3(256), 0, st>>>(c.corrMid, c.t4b, corrOut,
                                                       nullptr, CORR_N);
}

extern "C" void kernel_launch(void* const* d_in, const int* in_sizes, int n_in,
                              void* d_out, int out_size, void* d_ws, size_t ws_size,
                              hipStream_t stream) {
  (void)in_sizes; (void)out_size; (void)ws_size;
  Ctx c; c.st = stream;
  for (int i = 0; i < 68 && i < n_in; ++i) c.in[i] = (const float*)d_in[i];
  float* out = (float*)d_out;
  float* dout_corr = out;
  float* dout_src = out + CORR_N;
  float* dout_trg = out + CORR_N + FEAT_N;

  char* base = (char*)d_ws;
  size_t off = 0;
  auto af = [&](size_t n) { float* p = (float*)(base + off);
                            off += ((n * sizeof(float) + 255) & ~(size_t)255); return p; };
  auto ah = [&](size_t n) { half_t* p = (half_t*)(base + off);
                            off += ((n * sizeof(half_t) + 255) & ~(size_t)255); return p; };

  // fp16 weight copies
  c.wq = ah(327680);  c.wk = ah(327680);  c.wv = ah(65536);  c.wvx = ah(65536);
  c.l1w = ah(262144); c.l2w = ah(262144); c.xl1w = ah(262144); c.xl2w = ah(262144);
  half_t* pvpc = ah(16 * 352);
  half_t* pmc0 = ah(16 * 352);  half_t* pmc1 = ah(16 * 1312);
  half_t* pmr0 = ah(16 * 352);  half_t* pmr1 = ah(16 * 1312);
  half_t* pm2a = ah(16 * 352);  half_t* pm2b = ah(16 * 1312);
  half_t* pf1  = ah(16 * 96);   half_t* pf2  = ah(16 * 96);
  // fp16 activations
  half_t* corr_h  = ah(CORR_N);
  half_t* corrT_h = ah(CORR_N);
  c.lnA = ah(FEAT_N);
  c.cf = ah((size_t)NB * 256 * 1280);
  c.Qh = ah(FEAT_N); c.Kt = ah(FEAT_N); c.Vft = ah(FEAT_N);
  c.Vct = ah(CORR_N);
  c.KVtf = ah(262144); c.KVtc = ah(1048576);
  c.corrMid_h = ah(CORR_N);
  c.hidact = ah((size_t)NB * 256 * 1024);
  c.hid16h = ah((size_t)NB * 16 * 65536);
  c.corrR2h = ah(CORR_N); c.corrR4h = ah(CORR_N);
  c.Prow = ah(CORR_N); c.Pcol = ah(CORR_N);
  c.vSt = ah(FEAT_N); c.vTt = ah(FEAT_N);
  c.snorm = ah(FEAT_N); c.tnorm = ah(FEAT_N);
  c.tok_h = ah(FEAT_N);
  // fp32 scratch
  float* corrT_f = af(CORR_N);
  c.qraw = af(FEAT_N); c.kraw = af(FEAT_N); c.vraw = af(FEAT_N);
  c.Ksum = af(4096); c.Zb = af(16384);
  c.msgf = af(FEAT_N); c.msgc = af(CORR_N);
  c.featMid = af(FEAT_N); c.corrMid = af(CORR_N);
  c.hid = af((size_t)NB * 256 * 1024);
  c.t4a = af(CORR_N); c.t4b = af(CORR_N);
  c.hid16 = af((size_t)NB * 16 * 65536);
  float* c_src  = af(CORR_N);
  float* c_trgT = af(CORR_N);
  float* sfA = af(FEAT_N); float* tfA = af(FEAT_N);
  float* sfB = af(FEAT_N); float* tfB = af(FEAT_N);
  c.vS_f = af(FEAT_N); c.vT_f = af(FEAT_N);
  c.tok = af(FEAT_N);
  float* corrR1 = af(CORR_N); float* corrR2 = af(CORR_N);
  float* corrR3 = af(CORR_N); float* corrR4 = af(CORR_N);
  c.acc = af(64);

  hipStream_t st = stream;
  // ---- prep: fp16 weights + padded conv weights ----
  k_cvt<<<blocks_for(327680, 256), dim3(256), 0, st>>>(c.in[57], c.wq, 327680);
  k_cvt<<<blocks_for(327680, 256), dim3(256), 0, st>>>(c.in[10], c.wk, 327680);
  k_cvt<<<blocks_for(65536, 256), dim3(256), 0, st>>>(c.in[61], c.wv, 65536);
  k_cvt<<<blocks_for(65536, 256), dim3(256), 0, st>>>(c.in[59], c.wvx, 65536);
  k_cvt<<<blocks_for(262144, 256), dim3(256), 0, st>>>(c.in[14], c.l1w, 262144);
  k_cvt<<<blocks_for(262144, 256), dim3(256), 0, st>>>(c.in[16], c.l2w, 262144);
  k_cvt<<<blocks_for(262144, 256), dim3(256), 0, st>>>(c.in[28], c.xl1w, 262144);
  k_cvt<<<blocks_for(262144, 256), dim3(256), 0, st>>>(c.in[30], c.xl2w, 262144);
  k_prep_convw<<<blocks_for(16 * 352, 256), dim3(256), 0, st>>>(c.in[65], pvpc, 4, 4, 352);
  k_prep_convw<<<blocks_for(16 * 352, 256), dim3(256), 0, st>>>(c.in[20], pmc0, 4, 16, 352);
  k_prep_convw<<<blocks_for(16 * 1312, 256), dim3(256), 0, st>>>(c.in[24], pmc1, 16, 4, 1312);
  k_prep_convw<<<blocks_for(16 * 352, 256), dim3(256), 0, st>>>(c.in[34], pmr0, 4, 16, 352);
  k_prep_convw<<<blocks_for(16 * 1312, 256), dim3(256), 0, st>>>(c.in[38], pmr1, 16, 4, 1312);
  k_prep_convw<<<blocks_for(16 * 352, 256), dim3(256), 0, st>>>(c.in[42], pm2a, 4, 16, 352);
  k_prep_convw<<<blocks_for(16 * 1312, 256), dim3(256), 0, st>>>(c.in[46], pm2b, 16, 4, 1312);
  k_prep_convw<<<blocks_for(16 * 96, 256), dim3(256), 0, st>>>(c.in[4], pf1, 1, 4, 96);
  k_prep_convw<<<blocks_for(16 * 96, 256), dim3(256), 0, st>>>(c.in[8], pf2, 1, 4, 96);
  c.Lvpc = ConvLayer{pvpc, c.in[62], c.in[64], c.in[63], 4, 4, 352};
  c.Lmc0 = ConvLayer{pmc0, c.in[17], c.in[19], c.in[18], 4, 16, 352};
  c.Lmc1 = ConvLayer{pmc1, c.in[21], c.in[23], c.in[22], 16, 4, 1312};
  c.Lmr0 = ConvLayer{pmr0, c.in[31], c.in[33], c.in[32], 4, 16, 352};
  c.Lmr1 = ConvLayer{pmr1, c.in[35], c.in[37], c.in[36], 16, 4, 1312};
  c.Lm2a = ConvLayer{pm2a, c.in[39], c.in[41], c.in[40], 4, 16, 352};
  c.Lm2b = ConvLayer{pm2b, c.in[43], c.in[45], c.in[44], 16, 4, 1312};
  c.Lf1  = ConvLayer{pf1,  c.in[1],  c.in[3],  c.in[2],  1, 4, 96};
  c.Lf2  = ConvLayer{pf2,  c.in[5],  c.in[7],  c.in[6],  1, 4, 96};
  k_cvt<<<blocks_for(CORR_N, 256), dim3(256), 0, st>>>(c.in[0], corr_h, CORR_N);

  // ---- forward_attention, src branch ----
  run_fa(c, c.in[0], corr_h, c.in[66], c_src, sfA);
  // ---- forward_attention, trg branch (transposed corr domain) ----
  k_transpose_st<<<blocks_for(CORR_N, 256), dim3(256), 0, st>>>(c.in[0], corrT_f, corrT_h);
  run_fa(c, corrT_f, corrT_h, c.in[67], c_trgT, tfA);
  // corr_r = corr_src + corr_trg^T
  k_add_transposed<<<blocks_for(CORR_N, 256), dim3(256), 0, st>>>(c_src, c_trgT, corrR1);

  // ---- correlation token 1 + feat_to_corr1 encoder ----
  k_colnorm<<<dim3(NB * 256), dim3(256), 0, st>>>(sfA, c.snorm);
  k_colnorm<<<dim3(NB * 256), dim3(256), 0, st>>>(tfA, c.tnorm);
  launch_gemm(st, gp(c.snorm, 65536, 0, 256, c.tnorm, 65536, 0, 256, nullptr,
                     nullptr, 0, 0, 0, c.tok, 65536, 0, 256,
                     c.tok_h, 65536, 0, 256, 256, 1), 256, 256, NB);
  conv_gn_relu(c, c.Lf1, c.tok_h, c.t4a, c.t4b, nullptr);
  k_add<<<blocks_for(CORR_N, 256), dim3(256), 0, st>>>(corrR1, c.t4b, corrR2,
                                                       c.corrR2h, CORR_N);
  // ---- mlp_refine_corr encoder ----
  conv_gn_relu(c, c.Lmr0, c.corrR2h, c.hid16, nullptr, c.hid16h);
  conv_gn_relu(c, c.Lmr1, c.hid16h, c.t4a, c.t4b, nullptr);
  k_add<<<blocks_for(CORR_N, 256), dim3(256), 0, st>>>(corrR2, c.t4b, corrR3,
                                                       nullptr, CORR_N);

  // ---- forward_cross ----
  k_layernorm<<<dim3(NB * 256), dim3(256), 0, st>>>(sfA, c.in[52], c.in[51], c.lnA);
  launch_gemm(st, gp(c.lnA, 0, 0, 256, c.wvx, 0, 0, 256, c.in[58],
                     nullptr, 0, 0, 0, c.vS_f, 0, 0, 256,
                     nullptr, 0, 0, 0, 256, 1), 4096, 256, 1);
  k_headT<<<blocks_for(FEAT_N, 256), dim3(256), 0, st>>>(c.vS_f, c.vSt);
  k_layernorm<<<dim3(NB * 256), dim3(256), 0, st>>>(tfA, c.in[52], c.in[51], c.lnA);
  launch_gemm(st, gp(c.lnA, 0, 0, 256, c.wvx, 0, 0, 256, c.in[58],
                     nullptr, 0, 0, 0, c.vT_f, 0, 0, 256,
                     nullptr, 0, 0, 0, 256, 1), 4096, 256, 1);
  k_headT<<<blocks_for(FEAT_N, 256), dim3(256), 0, st>>>(c.vT_f, c.vTt);
  k_softmax_row<<<dim3(NB * NHD * 256), dim3(256), 0, st>>>(corrR3, c.Prow);
  k_softmax_col<<<dim3(NB * NHD * 256), dim3(256), 0, st>>>(corrR3, c.Pcol);
  // src_attn / trg_attn (per b,h), residual fused
  launch_gemm(st, gp(c.Prow, 262144, 65536, 256, c.vTt, 65536, 16384, 256, nullptr,
                     sfA, 65536, 64, 256, sfB, 65536, 64, 256,
                     nullptr, 0, 0, 0, 256, 4), 256, 64, 64);
  launch_gemm(st, gp(c.Pcol, 262144, 65536, 256, c.vSt, 65536, 16384, 256, nullptr,
                     tfA, 65536, 64, 256, tfB, 65536, 64, 256,
                     nullptr, 0, 0, 0, 256, 4), 256, 64, 64);
  // mlp_cross -> final src/trg features (written straight to d_out)
  run_mlp(c, c.in[53], c.in[54], c.xl1w, c.in[27], c.in[26], c.in[25],
          c.xl2w, c.in[29], sfB, dout_src);
  run_mlp(c, c.in[53], c.in[54], c.xl1w, c.in[27], c.in[26], c.in[25],
          c.xl2w, c.in[29], tfB, dout_trg);

  // ---- correlation token 2 + feat_to_corr2 encoder ----
  k_colnorm<<<dim3(NB * 256), dim3(256), 0, st>>>(dout_src, c.snorm);
  k_colnorm<<<dim3(NB * 256), dim3(256), 0, st>>>(dout_trg, c.tnorm);
  launch_gemm(st, gp(c.snorm, 65536, 0, 256, c.tnorm, 65536, 0, 256, nullptr,
                     nullptr, 0, 0, 0, c.tok, 65536, 0, 256,
                     c.tok_h, 65536, 0, 256, 256, 1), 256, 256, NB);
  conv_gn_relu(c, c.Lf2, c.tok_h, c.t4a, c.t4b, nullptr);
  k_add<<<blocks_for(CORR_N, 256), dim3(256), 0, st>>>(corrR3, c.t4b, corrR4,
                                                       c.corrR4h, CORR_N);
  // ---- mlp_refine_corr2 encoder -> final corr ----
  conv_gn_relu(c, c.Lm2a, c.corrR4h, c.hid16, nullptr, c.hid16h);
  conv_gn_relu(c, c.Lm2b, c.hid16h, c.t4a, c.t4b, nullptr);
  k_add<<<blocks_for(CORR_N, 256), dim3(256), 0, st>>>(corrR4, c.t4b, dout_corr,
                                                       nullptr, CORR_N);
}
